// BertAttention_8546984919540
// MI455X (gfx1250) — matrix operand order repaired and checked by reading, not compiled, not run on hardware
//
#include <hip/hip_runtime.h>
#include <hip/hip_bf16.h>
#include <hip/hip_fp16.h>

typedef __attribute__((ext_vector_type(16))) _Float16     v16h;
typedef __attribute__((ext_vector_type(8)))  _Float16     v8h;
typedef __attribute__((ext_vector_type(8)))  float        v8f;
typedef __attribute__((ext_vector_type(4)))  unsigned int u32x4;
typedef __attribute__((ext_vector_type(8)))  int          i32x8;
typedef __attribute__((ext_vector_type(4)))  int          i32x4;

#define NB    16
#define SQ    512
#define SKK   512
#define DD    768
#define NH    12
#define HD    64
#define NROW  (NB * SQ)      // 8192

#if __has_builtin(__builtin_amdgcn_tensor_load_to_lds) && \
    __has_builtin(__builtin_amdgcn_s_wait_tensorcnt)
#define HAVE_TDM 1
#else
#define HAVE_TDM 0
#endif

union FragU { v16h v; v8h h2[2]; };

// ---------------- fp32 -> f16 convert (coalesced) ----------------
__global__ void cvt_h(const float* __restrict__ in, _Float16* __restrict__ out, int n) {
    int i = blockIdx.x * blockDim.x + threadIdx.x;
    if (i < n) out[i] = (_Float16)in[i];
}

// ---------- fp32 -> f16 convert + transpose (W[k][n] -> Wt[n][k]) ----------
__global__ void cvt_t(const float* __restrict__ in, _Float16* __restrict__ out) {
    int i = blockIdx.x * blockDim.x + threadIdx.x;  // i = k*768 + n
    if (i < DD * DD) {
        int k = i / DD, n = i % DD;
        out[n * DD + k] = (_Float16)in[i];
    }
}

// ---------------- projection GEMM: C[M,768] = A[M,768] @ Wt^T + b ----------------
// Wt is N-major: Wt[n*768 + k].  Each wave computes a 32x32 tile (2x2 WMMA tiles),
// reusing each A/B fragment twice -> 4 WMMAs per 8 x 16B loads.
__global__ __launch_bounds__(256) void gemm_proj(
    const _Float16* __restrict__ A, const _Float16* __restrict__ Bt,
    const float* __restrict__ bias, _Float16* __restrict__ C, float scale)
{
    const int lane = threadIdx.x & 31;
    const int wave = threadIdx.x >> 5;
    const int hh   = lane >> 4;          // lane half (0/1)
    const int l16  = lane & 15;
    const int row0 = blockIdx.x * 32;
    const int col0 = blockIdx.y * 256 + wave * 32;

    const _Float16* arow0 = A  + (size_t)(row0 + l16) * DD;
    const _Float16* arow1 = arow0 + (size_t)16 * DD;
    const _Float16* brow0 = Bt + (size_t)(col0 + l16) * DD;
    const _Float16* brow1 = brow0 + (size_t)16 * DD;

    v8f acc00 = {}, acc01 = {}, acc10 = {}, acc11 = {};
    #pragma unroll 4
    for (int k = 0; k < DD; k += 32) {
        FragU a0, a1, b0, b1;
        // A-frag (ISA 16-bit A 16x32): halves 0-7 = K{0..7|8..15}, 8-15 = K{16..23|24..31}
        a0.h2[0] = *(const v8h*)(arow0 + k + hh * 8);
        a0.h2[1] = *(const v8h*)(arow0 + k + 16 + hh * 8);
        a1.h2[0] = *(const v8h*)(arow1 + k + hh * 8);
        a1.h2[1] = *(const v8h*)(arow1 + k + 16 + hh * 8);
        // B-frag: same K order as A-frag: halves 0-7 = K{0..7|8..15}, 8-15 = K{16..23|24..31}
        b0.h2[0] = *(const v8h*)(brow0 + k + hh * 8);
        b0.h2[1] = *(const v8h*)(brow0 + k + 16 + hh * 8);
        b1.h2[0] = *(const v8h*)(brow1 + k + hh * 8);
        b1.h2[1] = *(const v8h*)(brow1 + k + 16 + hh * 8);
        acc00 = __builtin_amdgcn_wmma_f32_16x16x32_f16(false, a0.v, false, b0.v, (short)0, acc00, false, false);
        acc01 = __builtin_amdgcn_wmma_f32_16x16x32_f16(false, a0.v, false, b1.v, (short)0, acc01, false, false);
        acc10 = __builtin_amdgcn_wmma_f32_16x16x32_f16(false, a1.v, false, b0.v, (short)0, acc10, false, false);
        acc11 = __builtin_amdgcn_wmma_f32_16x16x32_f16(false, a1.v, false, b1.v, (short)0, acc11, false, false);
    }
    const int   c0 = col0 + l16, c1 = col0 + 16 + l16;
    const float bv0 = bias[c0], bv1 = bias[c1];
    #pragma unroll
    for (int r = 0; r < 8; ++r) {
        int r0 = row0 + r + 8 * hh;           // C-frag row mapping, tile m=0
        int r1 = r0 + 16;                     // tile m=1
        C[(size_t)r0 * DD + c0] = (_Float16)((acc00[r] + bv0) * scale);
        C[(size_t)r0 * DD + c1] = (_Float16)((acc01[r] + bv1) * scale);
        C[(size_t)r1 * DD + c0] = (_Float16)((acc10[r] + bv0) * scale);
        C[(size_t)r1 * DD + c1] = (_Float16)((acc11[r] + bv1) * scale);
    }
}

// ---------------- fused attention: softmax(Q K^T) V, flash schedule ----------------
// grid.x = B*H, grid.y = SQ/128, block = 256 (8 waves x 16 query rows)
__global__ __launch_bounds__(256) void attn(
    const _Float16* __restrict__ Qh,   // [NROW, 768], already scaled by 1/8
    const _Float16* __restrict__ Kh,   // [NROW, 768]
    const _Float16* __restrict__ Vh,   // [NROW, 768]
    float* __restrict__ Out)           // [NB, SQ, 768]
{
    extern __shared__ _Float16 smem[];
    _Float16* sK  = smem;                    // [512][64]   row-major keys (this head)
    _Float16* sVt = smem + SKK * HD;         // [64][512]   V transposed
    _Float16* sP  = smem + 2 * SKK * HD;     // [8][16][32] per-wave P scratch

    const int bh   = blockIdx.x;
    const int b    = bh / NH, h = bh % NH;
    const int tid  = threadIdx.x;
    const int lane = tid & 31, wave = tid >> 5;
    const int hh   = lane >> 4, l16 = lane & 15;

#if HAVE_TDM
    // ---- stage K via the Tensor Data Mover: 2D tile 512x64 halves, row stride 768 ----
    if (wave == 0) {
        unsigned long long gaddr =
            (unsigned long long)(Kh + (size_t)(b * SKK) * DD + h * HD);
        u32x4 g0;
        g0[0] = 1u;                                   // count=1, user mode, no gather
        g0[1] = 0u;                                   // lds_addr = offset of sK (bytes)
        g0[2] = (unsigned)(gaddr & 0xffffffffu);      // global_addr[31:0]
        g0[3] = (unsigned)((gaddr >> 32) & 0x1ffffffu) | (2u << 30);  // [56:32] | type=2
        i32x8 g1;
        g1[0] = (int)(1u << 16);                      // data_size=1 (2 bytes/elem)
        g1[1] = (int)(64u << 16);                     // tensor_dim0 = 64  (bits 79:48)
        g1[2] = (int)(512u << 16);                    // tensor_dim1 = 512 (bits 111:80)
        g1[3] = (int)(64u << 16);                     // tile_dim0 = 64    (bits 127:112)
        g1[4] = 512;                                  // tile_dim1 = 512   (bits 143:128)
        g1[5] = 768;                                  // tensor_dim0_stride = 768
        g1[6] = 0;
        g1[7] = 0;
        i32x4 gz4 = {0, 0, 0, 0};
#if defined(__clang_major__) && (__clang_major__ >= 23)
        i32x8 gz8 = {0, 0, 0, 0, 0, 0, 0, 0};
        __builtin_amdgcn_tensor_load_to_lds(g0, g1, gz4, gz4, gz8, 0);
#else
        __builtin_amdgcn_tensor_load_to_lds(g0, g1, gz4, gz4, 0);
#endif
    }
    // ---- all waves stage V transposed meanwhile ----
    for (int it = tid; it < (SKK * HD) / 8; it += 256) {
        int key = it >> 3;
        int d   = (it & 7) << 3;
        v8h vv = *(const v8h*)(Vh + (size_t)(b * SKK + key) * DD + h * HD + d);
        #pragma unroll
        for (int e = 0; e < 8; ++e) sVt[(d + e) * SKK + key] = vv[e];
    }
    if (wave == 0) __builtin_amdgcn_s_wait_tensorcnt(0);
#else
    for (int it = tid; it < (SKK * HD) / 8; it += 256) {
        int key = it >> 3;
        int d   = (it & 7) << 3;
        size_t g = (size_t)(b * SKK + key) * DD + h * HD + d;
        v8h kv = *(const v8h*)(Kh + g);
        *(v8h*)(sK + key * HD + d) = kv;
        v8h vv = *(const v8h*)(Vh + g);
        #pragma unroll
        for (int e = 0; e < 8; ++e) sVt[(d + e) * SKK + key] = vv[e];
    }
#endif
    __syncthreads();

    // ---- Q fragments for this wave's 16 rows (K-dim = 64 -> 2 frags) ----
    const int qrow = blockIdx.y * 128 + wave * 16 + l16;
    const _Float16* qp = Qh + (size_t)(b * SQ + qrow) * DD + h * HD;
    FragU qa[2];
    #pragma unroll
    for (int t = 0; t < 2; ++t) {
        qa[t].h2[0] = *(const v8h*)(qp + t * 32 + hh * 8);
        qa[t].h2[1] = *(const v8h*)(qp + t * 32 + 16 + hh * 8);
    }

    v8f o[4] = {{}, {}, {}, {}};
    float m[8], l[8];
    #pragma unroll
    for (int r = 0; r < 8; ++r) { m[r] = -1e30f; l[r] = 0.0f; }
    _Float16* myP = sP + wave * (16 * 32);

    // ---- stream over 32-key chunks (two 16-wide score tiles per chunk) ----
    for (int j2 = 0; j2 < SKK / 32; ++j2) {
        v8f s0 = {}, s1 = {};
        {
            const _Float16* kp0 = sK + ((2 * j2 + 0) * 16 + l16) * HD;
            const _Float16* kp1 = sK + ((2 * j2 + 1) * 16 + l16) * HD;
            #pragma unroll
            for (int t = 0; t < 2; ++t) {
                FragU kb0, kb1;
                kb0.h2[0] = *(const v8h*)(kp0 + t * 32 + hh * 8);
                kb0.h2[1] = *(const v8h*)(kp0 + t * 32 + 16 + hh * 8);
                kb1.h2[0] = *(const v8h*)(kp1 + t * 32 + hh * 8);
                kb1.h2[1] = *(const v8h*)(kp1 + t * 32 + 16 + hh * 8);
                s0 = __builtin_amdgcn_wmma_f32_16x16x32_f16(false, qa[t].v, false, kb0.v, (short)0, s0, false, false);
                s1 = __builtin_amdgcn_wmma_f32_16x16x32_f16(false, qa[t].v, false, kb1.v, (short)0, s1, false, false);
            }
        }
        // combined online-softmax update for all 32 keys of this chunk
        #pragma unroll
        for (int r = 0; r < 8; ++r) {
            float mx = fmaxf(s0[r], s1[r]);
            mx = fmaxf(mx, __shfl_xor(mx, 1));
            mx = fmaxf(mx, __shfl_xor(mx, 2));
            mx = fmaxf(mx, __shfl_xor(mx, 4));
            mx = fmaxf(mx, __shfl_xor(mx, 8));
            float mnew  = fmaxf(m[r], mx);
            float alpha = __expf(m[r] - mnew);
            float p0    = __expf(s0[r] - mnew);
            float p1    = __expf(s1[r] - mnew);
            float ps = p0 + p1;
            ps += __shfl_xor(ps, 1);
            ps += __shfl_xor(ps, 2);
            ps += __shfl_xor(ps, 4);
            ps += __shfl_xor(ps, 8);
            l[r] = l[r] * alpha + ps;
            m[r] = mnew;
            #pragma unroll
            for (int nt = 0; nt < 4; ++nt) o[nt][r] *= alpha;
            // park P in LDS (C-layout -> row-major 16x32); per-wave private scratch
            myP[(r + 8 * hh) * 32 + l16]      = (_Float16)p0;
            myP[(r + 8 * hh) * 32 + 16 + l16] = (_Float16)p1;
        }
        // same-wave LDS ops are in-order (DScnt); fence the compiler only.
        __asm__ __volatile__("" ::: "memory");

        // A-frag of P (16x32) and 4 PV WMMAs over the 64 head dims
        FragU pa;
        const _Float16* pp = myP + l16 * 32;
        pa.h2[0] = *(const v8h*)(pp + hh * 8);
        pa.h2[1] = *(const v8h*)(pp + 16 + hh * 8);
        #pragma unroll
        for (int nt = 0; nt < 4; ++nt) {
            const _Float16* vp = sVt + (nt * 16 + l16) * SKK + j2 * 32;
            FragU vb;
            vb.h2[0] = *(const v8h*)(vp + hh * 8);
            vb.h2[1] = *(const v8h*)(vp + 16 + hh * 8);
            o[nt] = __builtin_amdgcn_wmma_f32_16x16x32_f16(false, pa.v, false, vb.v, (short)0, o[nt], false, false);
        }
        __asm__ __volatile__("" ::: "memory");
    }

    // ---- normalize and store [B, SQ, H*HD] fp32 ----
    #pragma unroll
    for (int r = 0; r < 8; ++r) {
        float inv = 1.0f / l[r];
        int row = blockIdx.y * 128 + wave * 16 + r + 8 * hh;
        float* op = Out + (size_t)(b * SQ + row) * DD + h * HD;
        #pragma unroll
        for (int nt = 0; nt < 4; ++nt) op[nt * 16 + l16] = o[nt][r] * inv;
    }
}

extern "C" void kernel_launch(void* const* d_in, const int* in_sizes, int n_in,
                              void* d_out, int out_size, void* d_ws, size_t ws_size,
                              hipStream_t stream) {
    (void)in_sizes; (void)n_in; (void)out_size; (void)ws_size;
    const float* hs  = (const float*)d_in[0];
    const float* ctx = (const float*)d_in[1];
    const float* Wq  = (const float*)d_in[2];
    const float* bq  = (const float*)d_in[3];
    const float* Wk  = (const float*)d_in[4];
    const float* bk  = (const float*)d_in[5];
    const float* Wv  = (const float*)d_in[6];
    const float* bv  = (const float*)d_in[7];
    float* out = (float*)d_out;

    // workspace layout (halves)
    _Float16* ws    = (_Float16*)d_ws;
    const size_t NM = (size_t)NROW * DD;     // 6,291,456
    const size_t NW = (size_t)DD * DD;       //   589,824
    _Float16* hs_h  = ws;
    _Float16* ctx_h = hs_h  + NM;
    _Float16* wqt   = ctx_h + NM;
    _Float16* wkt   = wqt + NW;
    _Float16* wvt   = wkt + NW;
    _Float16* qh    = wvt + NW;
    _Float16* kh    = qh + NM;
    _Float16* vh    = kh + NM;

    // 1) convert activations to f16
    {
        int n = (int)NM;
        int blocks = (n + 255) / 256;
        cvt_h<<<blocks, 256, 0, stream>>>(hs,  hs_h,  n);
        cvt_h<<<blocks, 256, 0, stream>>>(ctx, ctx_h, n);
    }
    // 2) convert + transpose weights
    {
        int blocks = ((int)NW + 255) / 256;
        cvt_t<<<blocks, 256, 0, stream>>>(Wq, wqt);
        cvt_t<<<blocks, 256, 0, stream>>>(Wk, wkt);
        cvt_t<<<blocks, 256, 0, stream>>>(Wv, wvt);
    }
    // 3) Q/K/V projections (fold 1/sqrt(HD)=0.125 into Q)
    {
        dim3 grid(NROW / 32, DD / 256);
        gemm_proj<<<grid, 256, 0, stream>>>(hs_h,  wqt, bq, qh, 0.125f);
        gemm_proj<<<grid, 256, 0, stream>>>(ctx_h, wkt, bk, kh, 1.0f);
        gemm_proj<<<grid, 256, 0, stream>>>(ctx_h, wvt, bv, vh, 1.0f);
    }
    // 4) fused flash attention (8 waves, 128 Q rows per block)
    {
        int smem = (2 * SKK * HD + 8 * 16 * 32) * (int)sizeof(_Float16); // 136 KB
        (void)hipFuncSetAttribute((const void*)attn,
                                  hipFuncAttributeMaxDynamicSharedMemorySize, smem);
        dim3 grid(NB * NH, SQ / 128);
        attn<<<grid, 256, smem, stream>>>(qh, kh, vh, out);
    }
}
